// MultiHeadAttn_18545668784946
// MI455X (gfx1250) — compile-verified, matrix operand order checked by reading, not run on hardware
//
#include <hip/hip_runtime.h>

typedef __bf16 bf16_t;
typedef __attribute__((ext_vector_type(16))) __bf16 v16bf;
typedef __attribute__((ext_vector_type(8)))  __bf16 v8bf;
typedef __attribute__((ext_vector_type(8)))  float  v8f;
typedef int i128_t __attribute__((vector_size(16)));   // int4, matches builtin proto

#define DMODEL 1024
#define SEQ    2048
#define NHEADS 16
#define DK     64
#define ROWS   4096   // BATCH * SEQ

union FragU { v16bf v; v8bf h[2]; };

static __device__ __forceinline__ v16bf make_frag(v8bf lo, v8bf hi) {
  FragU u; u.h[0] = lo; u.h[1] = hi; return u.v;
}

static __device__ __forceinline__ v8f wmma_bf16(v16bf a, v16bf b, v8f c) {
  return __builtin_amdgcn_wmma_f32_16x16x32_bf16(false, a, false, b, (short)0, c, false, false);
}

// ---- CDNA5 async global->LDS staging (ASYNCcnt path), with portable fallback ----
#if __has_builtin(__builtin_amdgcn_global_load_async_to_lds_b128)
#define ASYNC_LDS 1
#else
#define ASYNC_LDS 0
#endif

static __device__ __forceinline__ void stage16(const bf16_t* g, bf16_t* l) {
#if ASYNC_LDS
  __builtin_amdgcn_global_load_async_to_lds_b128(
      (__attribute__((address_space(1))) i128_t*)const_cast<bf16_t*>(g),
      (__attribute__((address_space(3))) i128_t*)l, 0, 0);
#else
  *(v8bf*)l = *(const v8bf*)g;       // global_load_b128 + ds_store_b128
#endif
}

static __device__ __forceinline__ void wait_async_done() {
#if ASYNC_LDS
#if __has_builtin(__builtin_amdgcn_s_wait_asynccnt)
  __builtin_amdgcn_s_wait_asynccnt(0);
#else
  asm volatile("s_wait_asynccnt 0x0" ::: "memory");
#endif
#endif
}

// ---------------- conversion kernels ----------------

__global__ void cvt_f32_bf16(const float* __restrict__ in, bf16_t* __restrict__ out, int n) {
  int i = blockIdx.x * blockDim.x + threadIdx.x;
  if (i < n) out[i] = (bf16_t)in[i];
}

// in: [rows][cols] f32  ->  out: [cols][rows] bf16
__global__ void cvt_transpose(const float* __restrict__ in, bf16_t* __restrict__ out,
                              int rows, int cols) {
  int i = blockIdx.x * blockDim.x + threadIdx.x;
  if (i < rows * cols) {
    int r = i / cols, c = i % cols;
    out[(size_t)c * rows + r] = (bf16_t)in[i];
  }
}

// -------- tiled WMMA GEMM: 64x128 block tile, 2x2 WMMA register block/wave --------
// C[M x N] = A[M x K] * B[K x N], B supplied transposed: Bt[N x K].
// Block: 256 thr = 8 waves (2 x 4); wave owns 32x32 C (4 accumulators).
// K-step 32: async-stage A(64x32)+Bt(128x32) into LDS double buffers.

template<bool F32OUT, bool TRANS_OUT>
__global__ __launch_bounds__(256)
void gemm_bf16_wmma(const bf16_t* __restrict__ A, const bf16_t* __restrict__ Bt,
                    bf16_t* __restrict__ Cb, float* __restrict__ Cf,
                    int K, int ldc) {
  __shared__ bf16_t As[2][64 * 32];    // [buf][m][k]  4 KB per buf
  __shared__ bf16_t Bs[2][128 * 32];   // [buf][n][k]  8 KB per buf

  const int t    = threadIdx.x;
  const int lane = t & 31;
  const int wid  = t >> 5;
  const int l15  = lane & 15;
  const int g    = lane >> 4;
  const int g8   = g << 3;

  const int rowBase = blockIdx.y * 64;
  const int colBase = blockIdx.x * 128;
  const int wr = (wid >> 2) * 32;      // wave row offset in 64-row tile
  const int wc = (wid & 3) * 32;       // wave col offset in 128-col tile

  // staging: 16B chunks. A: 256 chunks (64 rows x 4); B: 512 chunks (128 rows x 4).
  const int sr  = t >> 2;              // 0..63
  const int sch = (t & 3) * 8;
  const bf16_t* aSrc  = A  + (size_t)(rowBase + sr) * K + sch;
  const bf16_t* bSrc0 = Bt + (size_t)(colBase + sr) * K + sch;
  const bf16_t* bSrc1 = Bt + (size_t)(colBase + 64 + sr) * K + sch;

  const int nk = K / 32;
  // prologue: stage tile 0 into buffer 0
  stage16(aSrc,  &As[0][sr * 32 + sch]);
  stage16(bSrc0, &Bs[0][sr * 32 + sch]);
  stage16(bSrc1, &Bs[0][(64 + sr) * 32 + sch]);

  v8f cc[2][2];
#pragma unroll
  for (int i = 0; i < 2; ++i)
#pragma unroll
    for (int j = 0; j < 2; ++j) { v8f z = {}; cc[i][j] = z; }

  for (int kt = 0; kt < nk; ++kt) {
    wait_async_done();
    __syncthreads();                 // current buffer fully in LDS

    if (kt + 1 < nk) {               // issue next tile while computing
      const int nb = (kt + 1) & 1;
      const int ko = (kt + 1) * 32;
      stage16(aSrc + ko,  &As[nb][sr * 32 + sch]);
      stage16(bSrc0 + ko, &Bs[nb][sr * 32 + sch]);
      stage16(bSrc1 + ko, &Bs[nb][(64 + sr) * 32 + sch]);
    }

    const int cb = kt & 1;
    const bf16_t* Ab = &As[cb][0];
    const bf16_t* Bb = &Bs[cb][0];
    v16bf a0 = make_frag(*(const v8bf*)(Ab + (wr + l15) * 32 + g8),
                         *(const v8bf*)(Ab + (wr + l15) * 32 + 16 + g8));
    v16bf a1 = make_frag(*(const v8bf*)(Ab + (wr + 16 + l15) * 32 + g8),
                         *(const v8bf*)(Ab + (wr + 16 + l15) * 32 + 16 + g8));
    v16bf b0 = make_frag(*(const v8bf*)(Bb + (wc + l15) * 32 + g8),
                         *(const v8bf*)(Bb + (wc + l15) * 32 + 16 + g8));
    v16bf b1 = make_frag(*(const v8bf*)(Bb + (wc + 16 + l15) * 32 + g8),
                         *(const v8bf*)(Bb + (wc + 16 + l15) * 32 + 16 + g8));

    cc[0][0] = wmma_bf16(a0, b0, cc[0][0]);
    cc[0][1] = wmma_bf16(a0, b1, cc[0][1]);
    cc[1][0] = wmma_bf16(a1, b0, cc[1][0]);
    cc[1][1] = wmma_bf16(a1, b1, cc[1][1]);

    __syncthreads();                 // reads done before buffer is rewritten
  }

#pragma unroll
  for (int i = 0; i < 2; ++i)
#pragma unroll
    for (int j = 0; j < 2; ++j)
#pragma unroll
      for (int r = 0; r < 8; ++r) {
        int row = rowBase + wr + i * 16 + r + 8 * g;
        int col = colBase + wc + j * 16 + l15;
        if constexpr (F32OUT) {
          Cf[(size_t)row * ldc + col] = cc[i][j][r];
        } else if constexpr (TRANS_OUT) {
          Cb[(size_t)col * ldc + row] = (bf16_t)cc[i][j][r];
        } else {
          Cb[(size_t)row * ldc + col] = (bf16_t)cc[i][j][r];
        }
      }
}

// ---------------- flash attention (causal) ----------------
// 1 wave = one (batch, head, 16-query tile). 32-key tiles streamed; no block sync
// (waves have divergent causal trip counts).
// Q,K: [ROWS][DMODEL] bf16 (head-major cols). Vt: [DMODEL][ROWS] bf16.
// O:   [ROWS][DMODEL] bf16.

__global__ __launch_bounds__(256)
void flash_attn_wmma(const bf16_t* __restrict__ Qm, const bf16_t* __restrict__ Kmat,
                     const bf16_t* __restrict__ Vt, bf16_t* __restrict__ O) {
  __shared__ bf16_t lds_p[8 * 16 * 32];   // per-wave 16x32 P staging (8 KB total)

  const int lane = threadIdx.x & 31;
  const int wid  = threadIdx.x >> 5;
  const int l15  = lane & 15;
  const int g    = lane >> 4;
  const int g8   = g << 3;
  bf16_t* pl = lds_p + wid * (16 * 32);

  const int w   = blockIdx.x * 8 + wid;        // 0..4095
  const int b   = w >> 11;
  const int rem = w & 2047;
  const int h   = rem >> 7;
  const int q0  = (rem & 127) << 4;
  const int bRow = b * SEQ;
  const int hOff = h * DK;

  const bf16_t* Qp = Qm + (size_t)(bRow + q0 + l15) * DMODEL + hOff;
  const v16bf qf0 = make_frag(*(const v8bf*)(Qp + g8),      *(const v8bf*)(Qp + 16 + g8));
  const v16bf qf1 = make_frag(*(const v8bf*)(Qp + 32 + g8), *(const v8bf*)(Qp + 48 + g8));

  float mrow[8], lsum[8];
  v8f acc[4];
#pragma unroll
  for (int r = 0; r < 8; ++r) { mrow[r] = -3.0e38f; lsum[r] = 0.0f; }
#pragma unroll
  for (int d = 0; d < 4; ++d) { v8f z = {}; acc[d] = z; }

  const float scale = 0.125f;                  // 1/sqrt(DK)
  const int qmax = q0 + 15;

  for (int j0 = 0; j0 <= qmax; j0 += 32) {
    // ---- scores: S(16x32) = Q(16x64) x K_tile^T ----
    v8f sc[2];
#pragma unroll
    for (int G = 0; G < 2; ++G) {
      const bf16_t* Kp = Kmat + (size_t)(bRow + j0 + 16 * G + l15) * DMODEL + hOff;
      v16bf b0 = make_frag(*(const v8bf*)(Kp + g8),      *(const v8bf*)(Kp + 16 + g8));
      v16bf b1 = make_frag(*(const v8bf*)(Kp + 32 + g8), *(const v8bf*)(Kp + 48 + g8));
      v8f s = {};
      s = wmma_bf16(qf0, b0, s);
      s = wmma_bf16(qf1, b1, s);
      sc[G] = s;
    }

    // ---- causal mask + scale ----
#pragma unroll
    for (int G = 0; G < 2; ++G) {
      int col = j0 + 16 * G + l15;
#pragma unroll
      for (int r = 0; r < 8; ++r) {
        int rowq = q0 + r + 8 * g;
        float v = sc[G][r] * scale;
        sc[G][r] = (col <= rowq) ? v : -3.0e38f;
      }
    }

    // ---- online softmax (each row lives in one 16-lane half) ----
#pragma unroll
    for (int r = 0; r < 8; ++r) {
      float t = fmaxf(sc[0][r], sc[1][r]);
#pragma unroll
      for (int off = 8; off >= 1; off >>= 1) t = fmaxf(t, __shfl_xor(t, off, 32));
      float newm  = fmaxf(mrow[r], t);
      float alpha = __expf(mrow[r] - newm);
      mrow[r] = newm;
      sc[0][r] = __expf(sc[0][r] - newm);
      sc[1][r] = __expf(sc[1][r] - newm);
      float s2 = sc[0][r] + sc[1][r];
#pragma unroll
      for (int off = 8; off >= 1; off >>= 1) s2 += __shfl_xor(s2, off, 32);
      lsum[r] = lsum[r] * alpha + s2;
#pragma unroll
      for (int d = 0; d < 4; ++d) acc[d][r] *= alpha;
    }

    // ---- repack P: C-layout -> A-layout via per-wave LDS ----
#pragma unroll
    for (int G = 0; G < 2; ++G)
#pragma unroll
      for (int r = 0; r < 8; ++r) {
        int M = r + 8 * g;
        pl[M * 32 + G * 16 + l15] = (bf16_t)sc[G][r];
      }
    asm volatile("s_wait_dscnt 0x0" ::: "memory");   // intra-wave LDS RAW
    v16bf pa = make_frag(*(const v8bf*)(pl + l15 * 32 + g8),
                         *(const v8bf*)(pl + l15 * 32 + 16 + g8));

    // ---- acc += P(16x32) x V_tile(32x64) ----
#pragma unroll
    for (int d = 0; d < 4; ++d) {
      const bf16_t* Vp = Vt + (size_t)(hOff + d * 16 + l15) * ROWS + bRow + j0;
      v16bf bv = make_frag(*(const v8bf*)(Vp + g8), *(const v8bf*)(Vp + 16 + g8));
      acc[d] = wmma_bf16(pa, bv, acc[d]);
    }
  }

  // ---- epilogue: O = acc / l ----
#pragma unroll
  for (int r = 0; r < 8; ++r) {
    float inv = 1.0f / lsum[r];
    int row = bRow + q0 + r + 8 * g;
#pragma unroll
    for (int d = 0; d < 4; ++d) {
      O[(size_t)row * DMODEL + hOff + d * 16 + l15] = (bf16_t)(acc[d][r] * inv);
    }
  }
}

// ---------------- launch ----------------

extern "C" void kernel_launch(void* const* d_in, const int* in_sizes, int n_in,
                              void* d_out, int out_size, void* d_ws, size_t ws_size,
                              hipStream_t stream) {
  const float* x  = (const float*)d_in[0];
  const float* Wq = (const float*)d_in[1];
  const float* Wk = (const float*)d_in[2];
  const float* Wv = (const float*)d_in[3];
  const float* Wo = (const float*)d_in[4];
  float* out = (float*)d_out;

  // workspace layout (bf16): ~48 MB total
  bf16_t* xb  = (bf16_t*)d_ws;
  bf16_t* wqT = xb  + (size_t)ROWS   * DMODEL;
  bf16_t* wkT = wqT + (size_t)DMODEL * DMODEL;
  bf16_t* wvT = wkT + (size_t)DMODEL * DMODEL;
  bf16_t* woT = wvT + (size_t)DMODEL * DMODEL;
  bf16_t* Qm  = woT + (size_t)DMODEL * DMODEL;
  bf16_t* Km  = Qm  + (size_t)ROWS   * DMODEL;
  bf16_t* Vt  = Km  + (size_t)ROWS   * DMODEL;
  bf16_t* Om  = Vt  + (size_t)ROWS   * DMODEL;

  const int nx = ROWS * DMODEL;       // 4M
  const int nw = DMODEL * DMODEL;     // 1M
  cvt_f32_bf16 <<<nx / 256, 256, 0, stream>>>(x, xb, nx);
  cvt_transpose<<<nw / 256, 256, 0, stream>>>(Wq, wqT, DMODEL, DMODEL);
  cvt_transpose<<<nw / 256, 256, 0, stream>>>(Wk, wkT, DMODEL, DMODEL);
  cvt_transpose<<<nw / 256, 256, 0, stream>>>(Wv, wvT, DMODEL, DMODEL);
  cvt_transpose<<<nw / 256, 256, 0, stream>>>(Wo, woT, DMODEL, DMODEL);

  dim3 gg(DMODEL / 128, ROWS / 64);   // (8, 64)
  gemm_bf16_wmma<false, false><<<gg, 256, 0, stream>>>(xb, wqT, Qm, nullptr, DMODEL, DMODEL);
  gemm_bf16_wmma<false, false><<<gg, 256, 0, stream>>>(xb, wkT, Km, nullptr, DMODEL, DMODEL);
  gemm_bf16_wmma<false, true ><<<gg, 256, 0, stream>>>(xb, wvT, Vt, nullptr, DMODEL, ROWS);

  flash_attn_wmma<<<(ROWS / 16) * (NHEADS) / 8, 256, 0, stream>>>(Qm, Km, Vt, Om);

  gemm_bf16_wmma<true, false><<<gg, 256, 0, stream>>>(Om, woT, nullptr, out, DMODEL, DMODEL);
}
